// GPT_80822694576477
// MI455X (gfx1250) — matrix operand order repaired and checked by reading, not compiled, not run on hardware
//
#include <hip/hip_runtime.h>
#include <hip/hip_bf16.h>
#include <math.h>

// GPT-2 forward, CDNA5 (gfx1250), wave32, bf16 WMMA (V_WMMA_F32_16X16X32_BF16).
// V=371, E=512, H=8, L=8, B=4, T=1024, D=64, M=B*T=4096.
//
// Operands are carried as packed bf16 pairs (u32 = {k even, k odd}) in WMMA
// fragment order: A as [row][k2], B pre-transposed as [n][k2]. GEMM panels are
// staged with GLOBAL_LOAD_ASYNC_TO_LDS_B128 (ASYNCcnt) into double-buffered
// LDS; fragments are 2x ds_load_b128 per lane; fp32->bf16 RNE conversion
// happens once at each producer via v_perm_b32.

#define V_  371
#define E_  512
#define H_  8
#define L_  8
#define B_  4
#define T_  1024
#define D_  64
#define M_  4096

typedef __attribute__((ext_vector_type(16))) __bf16        v16bf;
typedef __attribute__((ext_vector_type(8)))  float         v8f;
typedef __attribute__((ext_vector_type(8)))  unsigned int  v8u;

// ---- bf16 packing helpers (RNE + v_perm_b32 merge) -------------------------
static __device__ inline unsigned int rne(float f) {
  unsigned int u = __builtin_bit_cast(unsigned int, f);
  return u + 0x7FFFu + ((u >> 16) & 1u);
}
static __device__ inline unsigned int pack2(float lo, float hi) {
  return __builtin_amdgcn_perm(rne(hi), rne(lo), 0x07060302u);
}

// ---- async global->LDS copy (CDNA5, ASYNCcnt-tracked, no VGPR round-trip) --
static __device__ inline void async_cp16(void* lds, const void* g) {
  asm volatile("global_load_async_to_lds_b128 %0, %1, off"
               :: "v"((unsigned)(size_t)lds),
                  "v"((unsigned long long)(size_t)g)
               : "memory");
}
static __device__ inline void wait_async0(void) {
  asm volatile("s_wait_asynccnt 0" ::: "memory");
}
static __device__ inline void wait_async4(void) {
  asm volatile("s_wait_asynccnt 4" ::: "memory");
}

// ---- fragment builders from packed-pair LDS tiles --------------------------
// A tile [row][k2]: lane m = l&15, g = l>>4; VGPR v holds k2 = col0 + (v>=4?8:0) + g*4 + (v&3).
static __device__ inline v16bf frag_a_pk(const unsigned int* tile, int pitch,
                                         int lane, int col0) {
  const int m = lane & 15, g = lane >> 4;
  const unsigned int* r = tile + m * pitch + col0 + g * 4;
  uint4 lo = *reinterpret_cast<const uint4*>(r);
  uint4 hi = *reinterpret_cast<const uint4*>(r + 8);
  v8u p;
  p[0] = lo.x; p[1] = lo.y; p[2] = lo.z; p[3] = lo.w;
  p[4] = hi.x; p[5] = hi.y; p[6] = hi.z; p[7] = hi.w;
  return __builtin_bit_cast(v16bf, p);
}
// B tile transposed [n][k2]: lane n = l&15, g = l>>4; VGPR v holds k2 = col0 + (v>=4?8:0) + g*4 + (v&3).
static __device__ inline v16bf frag_b_pk(const unsigned int* tile, int pitch,
                                         int lane, int row0, int col0) {
  const int n = lane & 15, g = lane >> 4;
  const unsigned int* r = tile + (row0 + n) * pitch + col0 + g * 4;
  uint4 lo = *reinterpret_cast<const uint4*>(r);
  uint4 hi = *reinterpret_cast<const uint4*>(r + 8);
  v8u p;
  p[0] = lo.x; p[1] = lo.y; p[2] = lo.z; p[3] = lo.w;
  p[4] = hi.x; p[5] = hi.y; p[6] = hi.z; p[7] = hi.w;
  return __builtin_bit_cast(v16bf, p);
}
static __device__ inline v8f wmma_bf16(v16bf a, v16bf b, v8f c) {
  return __builtin_amdgcn_wmma_f32_16x16x32_bf16(false, a, false, b, (short)0, c, false, false);
}

// ---- weight pre-pack: W[K,N] fp32 -> Wp[n][k2] packed bf16 -----------------
__global__ __launch_bounds__(256) void pack_w_trans(
    const float* __restrict__ W, unsigned int* __restrict__ Wp, int N, int K2)
{
  const int k2 = blockIdx.x * 256 + threadIdx.x;   // K2 multiple of 256
  const int n  = blockIdx.y;
  const float lo = W[(size_t)(2 * k2)     * N + n];
  const float hi = W[(size_t)(2 * k2 + 1) * N + n];
  Wp[(size_t)n * K2 + k2] = pack2(lo, hi);
}
// wte^T head: Wp[n][k2] = pack(wte[n][2k2], wte[n][2k2+1]); rows >= Nsrc zero-padded.
__global__ __launch_bounds__(256) void pack_w_rows(
    const float* __restrict__ W, unsigned int* __restrict__ Wp, int Nsrc, int K2)
{
  const int k2 = threadIdx.x;                      // K2 == 256
  const int n  = blockIdx.y;
  unsigned int v = 0;
  if (n < Nsrc) {
    const float2 p = *reinterpret_cast<const float2*>(W + (size_t)n * (2 * K2) + 2 * k2);
    v = pack2(p.x, p.y);
  }
  Wp[(size_t)n * K2 + k2] = v;
}

// ---- embedding: x = wte[idx] + wpe ----------------------------------------
__global__ __launch_bounds__(128) void embed_kernel(
    const int* __restrict__ idx, const float* __restrict__ wte,
    const float* __restrict__ wpe, float* __restrict__ x)
{
  const int tok = blockIdx.x;
  const int t   = tok & (T_ - 1);
  const int row = idx[tok];
  for (int e = threadIdx.x; e < E_; e += 128)
    x[(size_t)tok * E_ + e] = wte[(size_t)row * E_ + e] + wpe[(size_t)t * E_ + e];
}

// ---- LayerNorm: one wave per token, emits packed bf16 [tok][E/2] -----------
__global__ __launch_bounds__(128) void ln_kernel(
    const float* __restrict__ x, const float* __restrict__ w,
    unsigned int* __restrict__ out16)
{
  const int wave = threadIdx.x >> 5, lane = threadIdx.x & 31;
  const int tok = blockIdx.x * 4 + wave;
  const float* xr = x + (size_t)tok * E_;
  float s = 0.f, ss = 0.f;
  float2 vals[8];
#pragma unroll
  for (int j = 0; j < 8; ++j) {
    const float2 v = *reinterpret_cast<const float2*>(xr + 2 * (lane + j * 32));
    vals[j] = v; s += v.x + v.y; ss += v.x * v.x + v.y * v.y;
  }
#pragma unroll
  for (int m = 16; m >= 1; m >>= 1) {
    s  += __shfl_xor(s, m, 32);
    ss += __shfl_xor(ss, m, 32);
  }
  const float mean = s * (1.f / E_);
  const float var  = ss * (1.f / E_) - mean * mean;
  const float inv  = rsqrtf(var + 1e-5f);
#pragma unroll
  for (int j = 0; j < 8; ++j) {
    const int e2 = lane + j * 32;
    const float2 wv = *reinterpret_cast<const float2*>(w + 2 * e2);
    out16[(size_t)tok * (E_ / 2) + e2] =
        pack2((vals[j].x - mean) * inv * wv.x, (vals[j].y - mean) * inv * wv.y);
  }
}

// ---- GEMM: act(A16[M,K] @ Bp[K,N]) -> fp32 (+Res) or packed bf16 -----------
// Double-buffered LDS, panels staged via async-to-LDS b128 copies.
// 128 threads = 4 waves; block tile 64x64; each wave a 32x32 quadrant.
__global__ __launch_bounds__(128) void gemm_bf16_wmma(
    const unsigned int* __restrict__ A16, const unsigned int* __restrict__ Bp,
    const float* __restrict__ Res, float* __restrict__ Cf,
    unsigned int* __restrict__ C16, int M, int N, int K, int act)
{
  __shared__ __align__(16) unsigned int Ap[2][64][20];
  __shared__ __align__(16) unsigned int Bt[2][64][20];

  const int tid = threadIdx.x;
  const int lane = tid & 31, wave = tid >> 5;
  const int bm = blockIdx.y * 64, bn = blockIdx.x * 64;
  const int wm = (wave >> 1) * 32, wn = (wave & 1) * 32;
  const int K2 = K >> 1;
  const int m  = tid >> 2, c4 = (tid & 3) * 4;     // this thread's copy slots
  const int m1 = m + 32;

  // per-thread source pointers (4 async b128 per panel: A m, A m+32, B m, B m+32)
  const unsigned int* pa0 = A16 + (size_t)(bm + m)  * K2 + c4;
  const unsigned int* pa1 = A16 + (size_t)(bm + m1) * K2 + c4;
  const unsigned int* pb0 = Bp  + (size_t)(bn + m)  * K2 + c4;
  const unsigned int* pb1 = Bp  + (size_t)(bn + m1) * K2 + c4;

  // prologue: fill buffer 0
  async_cp16(&Ap[0][m][c4],  pa0);
  async_cp16(&Ap[0][m1][c4], pa1);
  async_cp16(&Bt[0][m][c4],  pb0);
  async_cp16(&Bt[0][m1][c4], pb1);

  v8f acc[2][2] = {};
  const int nk = K2 >> 4;
  for (int it = 0; it < nk; ++it) {
    const int cur = it & 1;
    if (it + 1 < nk) {                             // prefetch next panel
      const int nb = cur ^ 1, ko = (it + 1) * 16;
      async_cp16(&Ap[nb][m][c4],  pa0 + ko);
      async_cp16(&Ap[nb][m1][c4], pa1 + ko);
      async_cp16(&Bt[nb][m][c4],  pb0 + ko);
      async_cp16(&Bt[nb][m1][c4], pb1 + ko);
      wait_async4();                               // oldest 4 (current) done
    } else {
      wait_async0();
    }
    __syncthreads();                               // current panel visible

    const v16bf af0 = frag_a_pk(&Ap[cur][wm][0],      20, lane, 0);
    const v16bf af1 = frag_a_pk(&Ap[cur][wm + 16][0], 20, lane, 0);
    const v16bf bf0 = frag_b_pk(&Bt[cur][0][0], 20, lane, wn,      0);
    const v16bf bf1 = frag_b_pk(&Bt[cur][0][0], 20, lane, wn + 16, 0);
    acc[0][0] = wmma_bf16(af0, bf0, acc[0][0]);
    acc[0][1] = wmma_bf16(af0, bf1, acc[0][1]);
    acc[1][0] = wmma_bf16(af1, bf0, acc[1][0]);
    acc[1][1] = wmma_bf16(af1, bf1, acc[1][1]);
    __syncthreads();                               // reads done before overwrite
  }

  const int g = lane >> 4, nn = lane & 15;         // D layout: row = g*8+v, col = nn
#pragma unroll
  for (int mt = 0; mt < 2; ++mt)
#pragma unroll
    for (int nt = 0; nt < 2; ++nt)
#pragma unroll
      for (int v = 0; v < 8; ++v) {
        const int row = bm + wm + mt * 16 + g * 8 + v;
        const int col = bn + wn + nt * 16 + nn;
        float o = acc[mt][nt][v];
        if (act) o = 0.5f * o * (1.f + erff(o * 0.70710678118f));   // exact GELU
        if (C16) {
          const float po = __shfl_xor(o, 1, 16);
          if (!(nn & 1))
            C16[(size_t)row * (N >> 1) + (col >> 1)] = pack2(o, po);
        } else if (col < N) {
          if (Res) o += Res[(size_t)row * N + col];
          Cf[(size_t)row * N + col] = o;
        }
      }
}

// ---- Flash-style causal attention (consumes/produces packed bf16) ----------
// grid (T/64, B*H), 128 thr = 4 waves; wave w owns q rows [qb*64+w*16, +16).
__global__ __launch_bounds__(128) void attn_kernel(
    const unsigned int* __restrict__ qkv16, unsigned int* __restrict__ y16)
{
  __shared__ __align__(16) unsigned int Kp[32][36];     // [k-row][d2]
  __shared__ __align__(16) unsigned int Vt[64][20];     // [d][k2] (V transposed)
  __shared__ __align__(16) unsigned int Qp[4][16][36];  // per-wave Q [q-row][d2]
  __shared__ __align__(16) unsigned int Pp[4][16][20];  // per-wave P [q-row][k2]

  const int tid = threadIdx.x, lane = tid & 31, wave = tid >> 5;
  const int bh = blockIdx.y, b = bh >> 3, h = bh & 7;
  const int qb = blockIdx.x;
  const int q0 = qb * 64 + wave * 16;
  const float scale = 0.125f;                           // 1/sqrt(64), applied to S
  const int RQ = 3 * E_ / 2;                            // qkv16 row stride (u32)
  const int hq = h * (D_ / 2);

  // stage Q: 16 rows x 8 uint4 via async copies (per-wave region)
#pragma unroll
  for (int i = 0; i < 4; ++i) {
    const int idx = i * 32 + lane;
    const int r = idx >> 3, c4 = (idx & 7) * 4;
    async_cp16(&Qp[wave][r][c4],
               qkv16 + (size_t)(b * T_ + q0 + r) * RQ + hq + c4);
  }
  wait_async0();
  const v16bf aq0 = frag_a_pk(&Qp[wave][0][0], 36, lane, 0);
  const v16bf aq1 = frag_a_pk(&Qp[wave][0][0], 36, lane, 16);

  const int g = lane >> 4, nn = lane & 15;
  float mrow[8], lrow[8];
#pragma unroll
  for (int v = 0; v < 8; ++v) { mrow[v] = -1e30f; lrow[v] = 0.f; }
  v8f oacc[4] = {};

  const int kend = qb * 64 + 64;                        // causal upper bound
  for (int kb = 0; kb < kend; kb += 32) {
    // K panel: 32 rows x 8 uint4, async straight copy
#pragma unroll
    for (int i = 0; i < 2; ++i) {
      const int idx = i * 128 + tid;
      const int r = idx >> 3, c4 = (idx & 7) * 4;
      async_cp16(&Kp[r][c4],
                 qkv16 + (size_t)(b * T_ + kb + r) * RQ + (E_ / 2) + hq + c4);
    }
    // V panel transposed: d-packed rows -> k-packed columns via 2x v_perm
#pragma unroll
    for (int i = 0; i < 4; ++i) {
      const int idx = i * 128 + tid;                    // 512 tasks: (k2, d-pair)
      const int k2 = idx >> 5, dp = idx & 31;
      const size_t base = (size_t)(b * T_ + kb + 2 * k2) * RQ + E_ + hq + dp;
      const unsigned int r0 = qkv16[base];
      const unsigned int r1 = qkv16[base + RQ];
      Vt[2 * dp][k2]     = __builtin_amdgcn_perm(r1, r0, 0x05040100u);
      Vt[2 * dp + 1][k2] = __builtin_amdgcn_perm(r1, r0, 0x07060302u);
    }
    wait_async0();
    __syncthreads();

    if (kb <= q0 + 15) {                                // wave-uniform causal skip
      v8f s0 = {}, s1 = {};
      s0 = wmma_bf16(aq0, frag_b_pk(&Kp[0][0], 36, lane, 0,  0),  s0);
      s0 = wmma_bf16(aq1, frag_b_pk(&Kp[0][0], 36, lane, 0,  16), s0);
      s1 = wmma_bf16(aq0, frag_b_pk(&Kp[0][0], 36, lane, 16, 0),  s1);
      s1 = wmma_bf16(aq1, frag_b_pk(&Kp[0][0], 36, lane, 16, 16), s1);

      float alpha[8];
#pragma unroll
      for (int v = 0; v < 8; ++v) {
        const int q = q0 + g * 8 + v;
        float x0 = (kb + nn      <= q) ? s0[v] * scale : -1e30f;
        float x1 = (kb + 16 + nn <= q) ? s1[v] * scale : -1e30f;
        s0[v] = x0; s1[v] = x1;
        float mx = fmaxf(x0, x1);
#pragma unroll
        for (int mm = 8; mm >= 1; mm >>= 1) mx = fmaxf(mx, __shfl_xor(mx, mm, 16));
        const float mnew = fmaxf(mrow[v], mx);
        alpha[v] = __expf(mrow[v] - mnew);
        mrow[v] = mnew;
      }
#pragma unroll
      for (int nt = 0; nt < 4; ++nt)
#pragma unroll
        for (int v = 0; v < 8; ++v) oacc[nt][v] *= alpha[v];

#pragma unroll
      for (int v = 0; v < 8; ++v) {
        const float p0 = __expf(s0[v] - mrow[v]);
        const float p1 = __expf(s1[v] - mrow[v]);
        float rs = p0 + p1;
#pragma unroll
        for (int mm = 8; mm >= 1; mm >>= 1) rs += __shfl_xor(rs, mm, 16);
        lrow[v] = lrow[v] * alpha[v] + rs;
        const float q0p = __shfl_xor(p0, 1, 16);
        const float q1p = __shfl_xor(p1, 1, 16);
        const int row = g * 8 + v;
        const unsigned int pk = (nn & 1) ? pack2(q1p, p1) : pack2(p0, q0p);
        const int k2 = (nn & 1) ? (8 + (nn >> 1)) : (nn >> 1);
        Pp[wave][row][k2] = pk;
      }
      const v16bf ap = frag_a_pk(&Pp[wave][0][0], 20, lane, 0);
      oacc[0] = wmma_bf16(ap, frag_b_pk(&Vt[0][0], 20, lane, 0,  0), oacc[0]);
      oacc[1] = wmma_bf16(ap, frag_b_pk(&Vt[0][0], 20, lane, 16, 0), oacc[1]);
      oacc[2] = wmma_bf16(ap, frag_b_pk(&Vt[0][0], 20, lane, 32, 0), oacc[2]);
      oacc[3] = wmma_bf16(ap, frag_b_pk(&Vt[0][0], 20, lane, 48, 0), oacc[3]);
    }
    __syncthreads();
  }

  // write attention output packed bf16 (consumed only as GEMM A operand)
#pragma unroll
  for (int nt = 0; nt < 4; ++nt)
#pragma unroll
    for (int v = 0; v < 8; ++v) {
      const int q = q0 + g * 8 + v;
      const float val = oacc[nt][v] / lrow[v];
      const float par = __shfl_xor(val, 1, 16);
      if (!(nn & 1)) {
        const int c2 = (h * D_ + nt * 16 + nn) >> 1;
        y16[(size_t)(b * T_ + q) * (E_ / 2) + c2] = pack2(val, par);
      }
    }
}

// ---- driver ----------------------------------------------------------------
extern "C" void kernel_launch(void* const* d_in, const int* in_sizes, int n_in,
                              void* d_out, int out_size, void* d_ws, size_t ws_size,
                              hipStream_t stream)
{
  (void)in_sizes; (void)n_in; (void)out_size; (void)ws_size;
  const int*   idx   = (const int*)  d_in[0];
  const float* wte   = (const float*)d_in[1];
  const float* wpe   = (const float*)d_in[2];
  const float* attw  = (const float*)d_in[3];  // [L,E,3E]
  const float* projw = (const float*)d_in[4];  // [L,E,E]
  const float* fcw   = (const float*)d_in[5];  // [L,E,4E]
  const float* fcpw  = (const float*)d_in[6];  // [L,4E,E]
  const float* ln1w  = (const float*)d_in[7];
  const float* ln2w  = (const float*)d_in[8];
  const float* lnfw  = (const float*)d_in[9];
  float* logits = (float*)d_out;

  // workspace carve-up (u32 units)
  unsigned int* w32 = (unsigned int*)d_ws;
  float*        x    = (float*)w32;                      // [M,E] fp32 residual
  unsigned int* h16  = w32 + (size_t)M_ * E_;            // [M,E/2]
  unsigned int* qkv16= h16  + (size_t)M_ * E_ / 2;       // [M,3E/2]
  unsigned int* y16  = qkv16+ (size_t)M_ * 3 * E_ / 2;   // [M,E/2]
  unsigned int* ff16 = y16  + (size_t)M_ * E_ / 2;       // [M,2E]
  unsigned int* wp   = ff16 + (size_t)M_ * 2 * E_;       // packed weights
  const size_t szAtt = (size_t)(3 * E_) * (E_ / 2);      // [1536][256]
  const size_t szPrj = (size_t)E_ * (E_ / 2);            // [512][256]
  const size_t szFc  = (size_t)(4 * E_) * (E_ / 2);      // [2048][256]
  const size_t szFcp = (size_t)E_ * (2 * E_);            // [512][1024]
  unsigned int* wAtt = wp;
  unsigned int* wPrj = wAtt + L_ * szAtt;
  unsigned int* wFc  = wPrj + L_ * szPrj;
  unsigned int* wFcp = wFc  + L_ * szFc;
  unsigned int* wVoc = wFcp + L_ * szFcp;                // [384][256], rows >=371 zero

  const dim3 blk(128);

  // one-time weight pre-pack (per launch; deterministic)
  for (int l = 0; l < L_; ++l) {
    pack_w_trans<<<dim3(1, 3 * E_), 256, 0, stream>>>(
        attw + (size_t)l * E_ * 3 * E_, wAtt + l * szAtt, 3 * E_, E_ / 2);
    pack_w_trans<<<dim3(1, E_), 256, 0, stream>>>(
        projw + (size_t)l * E_ * E_, wPrj + l * szPrj, E_, E_ / 2);
    pack_w_trans<<<dim3(1, 4 * E_), 256, 0, stream>>>(
        fcw + (size_t)l * E_ * 4 * E_, wFc + l * szFc, 4 * E_, E_ / 2);
    pack_w_trans<<<dim3(4, E_), 256, 0, stream>>>(
        fcpw + (size_t)l * 4 * E_ * E_, wFcp + l * szFcp, E_, 2 * E_);
  }
  pack_w_rows<<<dim3(1, 384), 256, 0, stream>>>(wte, wVoc, V_, E_ / 2);

  embed_kernel<<<M_, blk, 0, stream>>>(idx, wte, wpe, x);

  for (int l = 0; l < L_; ++l) {
    ln_kernel<<<M_ / 4, blk, 0, stream>>>(x, ln1w + (size_t)l * E_, h16);
    gemm_bf16_wmma<<<dim3((3 * E_) / 64, M_ / 64), blk, 0, stream>>>(     // -> qkv16 (packed)
        h16, wAtt + l * szAtt, nullptr, nullptr, qkv16, M_, 3 * E_, E_, 0);
    attn_kernel<<<dim3(T_ / 64, B_ * H_), blk, 0, stream>>>(qkv16, y16);
    gemm_bf16_wmma<<<dim3(E_ / 64, M_ / 64), blk, 0, stream>>>(           // x += y @ proj
        y16, wPrj + l * szPrj, x, x, nullptr, M_, E_, E_, 0);
    ln_kernel<<<M_ / 4, blk, 0, stream>>>(x, ln2w + (size_t)l * E_, h16);
    gemm_bf16_wmma<<<dim3((4 * E_) / 64, M_ / 64), blk, 0, stream>>>(     // -> gelu, packed
        h16, wFc + l * szFc, nullptr, nullptr, ff16, M_, 4 * E_, E_, 1);
    gemm_bf16_wmma<<<dim3(E_ / 64, M_ / 64), blk, 0, stream>>>(           // x += ff @ fcproj
        ff16, wFcp + l * szFcp, x, x, nullptr, M_, E_, 4 * E_, 0);
  }

  ln_kernel<<<M_ / 4, blk, 0, stream>>>(x, lnfw, h16);
  gemm_bf16_wmma<<<dim3((V_ + 63) / 64, M_ / 64), blk, 0, stream>>>(      // tied head
      h16, wVoc, nullptr, logits, nullptr, M_, V_, E_, 0);
}
